// SpikingSelfAttention_2018634629158
// MI455X (gfx1250) — matrix operand order repaired and checked by reading, not compiled, not run on hardware
//
#include <hip/hip_runtime.h>
#include <hip/hip_bf16.h>

typedef __bf16 bf16;
typedef __attribute__((ext_vector_type(8)))  __bf16 v8bf;
typedef __attribute__((ext_vector_type(16))) __bf16 v16bf;
typedef __attribute__((ext_vector_type(8)))  float   v8f;

constexpr int kE = 768;    // embed dim
constexpr int kH = 12;     // heads
constexpr int kD = 64;     // head dim
constexpr int kS = 1024;   // seq len
constexpr int kB = 8;      // batch
constexpr int kN = 8192;   // tokens = kB*kS
constexpr int kT = 10;     // time window

// ---- WMMA fragment helpers (CDNA5 16-bit layouts, ISA 7.12.2) -------------
// A 16x32 bf16: lane holds row (lane&15); per-lane data = K[half*8 .. +7] in
// VGPR0-3 and K[16+half*8 .. +7] in VGPR4-7 (half = lane>>4).
__device__ __forceinline__ v16bf afrag(const bf16* p, int kh) {
  v8bf lo = *(const v8bf*)(p + kh * 8);
  v8bf hi = *(const v8bf*)(p + 16 + kh * 8);
  return __builtin_shufflevector(lo, hi, 0,1,2,3,4,5,6,7,8,9,10,11,12,13,14,15);
}
// B 32x16 bf16: lane holds column (lane&15); per-lane data = K[half*8 .. +7]
// and K[16+half*8 .. +7] from a row of B^T (same K order as afrag).
__device__ __forceinline__ v16bf bfrag(const bf16* p, int kh) {
  v8bf lo = *(const v8bf*)(p + kh * 8);
  v8bf hi = *(const v8bf*)(p + 16 + kh * 8);
  return __builtin_shufflevector(lo, hi, 0,1,2,3,4,5,6,7,8,9,10,11,12,13,14,15);
}
__device__ __forceinline__ v8f wmma_bf(v16bf a, v16bf b, v8f c) {
  return __builtin_amdgcn_wmma_f32_16x16x32_bf16(false, a, false, b,
                                                 (short)0, c, false, false);
}
__device__ __forceinline__ float sigm(float x) {
  return 1.0f / (1.0f + __expf(-x));
}

// ---- 1) rate encode: inp[t][n][e] = sigmoid(10*(clip(x)*T - t)), bf16 -----
__global__ void k_encode(const float* __restrict__ x, bf16* __restrict__ inp) {
  size_t i = (size_t)blockIdx.x * 256 + threadIdx.x;
  if (i >= (size_t)kN * kE) return;
  float p = fminf(fmaxf(x[i], 0.0f), 1.0f);
  float base = 100.0f * p;  // SLOPE * T * p
#pragma unroll
  for (int t = 0; t < kT; ++t) {
    float s = 1.0f / (1.0f + __expf(10.0f * (float)t - base));
    inp[(size_t)t * kN * kE + i] = (bf16)s;
  }
}

// ---- 2) weights f32 -> bf16 (row-major kept: rows of W are rows of B^T) ---
__global__ void k_cvtw(const float* __restrict__ Wq, const float* __restrict__ Wk,
                       const float* __restrict__ Wv, const float* __restrict__ Wo,
                       bf16* __restrict__ w) {
  const size_t EE = (size_t)kE * kE;
  size_t i = (size_t)blockIdx.x * 256 + threadIdx.x;
  if (i >= 4 * EE) return;
  int sel = (int)(i / EE);
  size_t j = i % EE;
  const float* src = (sel == 0) ? Wq : (sel == 1) ? Wk : (sel == 2) ? Wv : Wo;
  w[i] = (bf16)src[j];
}

// ---- 3) fused QKV GEMM + LIF recurrence over T steps ----------------------
// grid (kN/128, kE/64, 3); 256 thr = 8 waves in 4x2; wave tile 32x32 (2x2 WMMA)
__global__ __launch_bounds__(256) void k_qkv(
    const bf16* __restrict__ inp, const bf16* __restrict__ w,
    const float* __restrict__ bq, const float* __restrict__ bk,
    const float* __restrict__ bv,
    bf16* __restrict__ Qo, bf16* __restrict__ Ko, bf16* __restrict__ Vt) {
  const int lane = threadIdx.x & 31;
  const int wave = threadIdx.x >> 5;
  const int kh = lane >> 4, l16 = lane & 15;
  const int M0 = blockIdx.x * 128 + (wave >> 1) * 32;
  const int N0 = blockIdx.y * 64 + (wave & 1) * 32;
  const int sel = blockIdx.z;
  const bf16* W = w + (size_t)sel * kE * kE;
  const float* bias = (sel == 0) ? bq : (sel == 1) ? bk : bv;
  const float bcol0 = bias[N0 + l16];
  const float bcol1 = bias[N0 + 16 + l16];

  v8f vst[2][2] = {};  // LIF membrane state, WMMA C layout
  v8f acc[2][2] = {};  // spike-count accumulator
  for (int t = 0; t < kT; ++t) {
    const bf16* A = inp + (size_t)t * kN * kE;
    const bf16* ar0 = A + (size_t)(M0 + l16) * kE;
    const bf16* ar1 = A + (size_t)(M0 + 16 + l16) * kE;
    const bf16* br0 = W + (size_t)(N0 + l16) * kE;
    const bf16* br1 = W + (size_t)(N0 + 16 + l16) * kE;
    v8f c[2][2] = {};
    for (int k0 = 0; k0 < kE; k0 += 32) {
      v16bf a0 = afrag(ar0 + k0, kh);
      v16bf a1 = afrag(ar1 + k0, kh);
      v16bf b0 = bfrag(br0 + k0, kh);
      v16bf b1 = bfrag(br1 + k0, kh);
      c[0][0] = wmma_bf(a0, b0, c[0][0]);
      c[0][1] = wmma_bf(a0, b1, c[0][1]);
      c[1][0] = wmma_bf(a1, b0, c[1][0]);
      c[1][1] = wmma_bf(a1, b1, c[1][1]);
    }
#pragma unroll
    for (int mi = 0; mi < 2; ++mi)
#pragma unroll
      for (int ni = 0; ni < 2; ++ni)
#pragma unroll
        for (int r = 0; r < 8; ++r) {
          float I = c[mi][ni][r] + (ni ? bcol1 : bcol0);
          float v = 0.5f * (vst[mi][ni][r] + I);       // v += (I - v)/TAU
          float s = sigm(10.0f * (v - 1.0f));          // surrogate spike
          vst[mi][ni][r] = v * (1.0f - s);             // multiplicative reset
          acc[mi][ni][r] += s;
        }
  }
#pragma unroll
  for (int mi = 0; mi < 2; ++mi)
#pragma unroll
    for (int ni = 0; ni < 2; ++ni)
#pragma unroll
      for (int r = 0; r < 8; ++r) {
        int m = M0 + mi * 16 + r + 8 * kh;  // token row
        int e = N0 + ni * 16 + l16;         // output feature
        int b = m >> 10, s = m & 1023;
        int h = e >> 6, d = e & 63;
        int bh = b * kH + h;
        bf16 val = (bf16)acc[mi][ni][r];
        if (sel == 0)      Qo[((size_t)bh * kS + s) * kD + d] = val;
        else if (sel == 1) Ko[((size_t)bh * kS + s) * kD + d] = val;
        else               Vt[((size_t)bh * kD + d) * kS + s] = val;  // V^T
      }
}

// ---- 4) flash attention: per-wave 16 query rows, 32-key blocks ------------
// grid (kS/128, kB*kH); 256 thr = 8 waves
__global__ __launch_bounds__(256) void k_attn(
    const bf16* __restrict__ Q, const bf16* __restrict__ K,
    const bf16* __restrict__ Vt, bf16* __restrict__ AO) {
  __shared__ __align__(128) bf16 pl[8][16][32];  // per-wave P restage tile
  const int lane = threadIdx.x & 31;
  const int wave = threadIdx.x >> 5;
  const int kh = lane >> 4, l16 = lane & 15;
  const int bh = blockIdx.y;
  const int b = bh / kH, h = bh % kH;
  const int q0 = blockIdx.x * 128 + wave * 16;
  const bf16* Qb = Q + (size_t)bh * kS * kD;
  const bf16* Kb = K + (size_t)bh * kS * kD;
  const bf16* Vb = Vt + (size_t)bh * kD * kS;

  v16bf qa[2];  // Q A-fragments for K-dim steps d=0..31, 32..63
  qa[0] = afrag(Qb + (size_t)(q0 + l16) * kD + 0, kh);
  qa[1] = afrag(Qb + (size_t)(q0 + l16) * kD + 32, kh);

  float mrow[8], lrow[8], facs[8];
#pragma unroll
  for (int r = 0; r < 8; ++r) { mrow[r] = -1e30f; lrow[r] = 0.0f; }
  v8f O[4] = {};

  for (int key0 = 0; key0 < kS; key0 += 32) {
    v8f c0 = {}, c1 = {};  // scores tiles, keys [key0..+15], [key0+16..+31]
#pragma unroll
    for (int j = 0; j < 2; ++j) {
      int ks = j * 32;
      c0 = wmma_bf(qa[j], bfrag(Kb + (size_t)(key0 + l16) * kD + ks, kh), c0);
      c1 = wmma_bf(qa[j], bfrag(Kb + (size_t)(key0 + 16 + l16) * kD + ks, kh), c1);
    }
#pragma unroll
    for (int r = 0; r < 8; ++r) {
      float s0 = c0[r] * 0.125f;  // 1/sqrt(64)
      float s1 = c1[r] * 0.125f;
      float mx = fmaxf(s0, s1);   // row max across 16 lanes of half-group
      mx = fmaxf(mx, __shfl_xor(mx, 1));
      mx = fmaxf(mx, __shfl_xor(mx, 2));
      mx = fmaxf(mx, __shfl_xor(mx, 4));
      mx = fmaxf(mx, __shfl_xor(mx, 8));
      float mnew = fmaxf(mrow[r], mx);
      float fac = __expf(mrow[r] - mnew);
      float p0 = __expf(s0 - mnew);
      float p1 = __expf(s1 - mnew);
      float rs = p0 + p1;
      rs += __shfl_xor(rs, 1);
      rs += __shfl_xor(rs, 2);
      rs += __shfl_xor(rs, 4);
      rs += __shfl_xor(rs, 8);
      lrow[r] = lrow[r] * fac + rs;
      mrow[r] = mnew;
      facs[r] = fac;
      // restage P from C-layout to row-major [qrow][key] for A-fragment use
      pl[wave][r + 8 * kh][l16] = (bf16)p0;
      pl[wave][r + 8 * kh][16 + l16] = (bf16)p1;
    }
#pragma unroll
    for (int di = 0; di < 4; ++di)
#pragma unroll
      for (int r = 0; r < 8; ++r) O[di][r] *= facs[r];
    v16bf pa = afrag(&pl[wave][l16][0], kh);  // same-wave DS: in-order
#pragma unroll
    for (int di = 0; di < 4; ++di)
      O[di] = wmma_bf(pa, bfrag(Vb + (size_t)(di * 16 + l16) * kS + key0, kh),
                      O[di]);
  }
#pragma unroll
  for (int di = 0; di < 4; ++di)
#pragma unroll
    for (int r = 0; r < 8; ++r) {
      float o = O[di][r] / lrow[r];
      int qrow = q0 + r + 8 * kh;
      int n = b * kS + qrow;
      int e = h * kD + di * 16 + l16;
      AO[(size_t)n * kE + e] = (bf16)o;
    }
}

// ---- 5) output projection + fresh-state LIF: sigmoid(10*(I/2 - 1)) --------
// grid (kN/128, kE/64); same tiling as k_qkv
__global__ __launch_bounds__(256) void k_oproj(
    const bf16* __restrict__ A, const bf16* __restrict__ Wo,
    const float* __restrict__ bo, float* __restrict__ out) {
  const int lane = threadIdx.x & 31;
  const int wave = threadIdx.x >> 5;
  const int kh = lane >> 4, l16 = lane & 15;
  const int M0 = blockIdx.x * 128 + (wave >> 1) * 32;
  const int N0 = blockIdx.y * 64 + (wave & 1) * 32;
  const float bcol0 = bo[N0 + l16];
  const float bcol1 = bo[N0 + 16 + l16];
  const bf16* ar0 = A + (size_t)(M0 + l16) * kE;
  const bf16* ar1 = A + (size_t)(M0 + 16 + l16) * kE;
  const bf16* br0 = Wo + (size_t)(N0 + l16) * kE;
  const bf16* br1 = Wo + (size_t)(N0 + 16 + l16) * kE;
  v8f c[2][2] = {};
  for (int k0 = 0; k0 < kE; k0 += 32) {
    v16bf a0 = afrag(ar0 + k0, kh);
    v16bf a1 = afrag(ar1 + k0, kh);
    v16bf b0 = bfrag(br0 + k0, kh);
    v16bf b1 = bfrag(br1 + k0, kh);
    c[0][0] = wmma_bf(a0, b0, c[0][0]);
    c[0][1] = wmma_bf(a0, b1, c[0][1]);
    c[1][0] = wmma_bf(a1, b0, c[1][0]);
    c[1][1] = wmma_bf(a1, b1, c[1][1]);
  }
#pragma unroll
  for (int mi = 0; mi < 2; ++mi)
#pragma unroll
    for (int ni = 0; ni < 2; ++ni)
#pragma unroll
      for (int r = 0; r < 8; ++r) {
        float I = c[mi][ni][r] + (ni ? bcol1 : bcol0);
        float s = sigm(10.0f * (0.5f * I - 1.0f));  // v = I/TAU from v0=0
        int m = M0 + mi * 16 + r + 8 * kh;
        int e = N0 + ni * 16 + l16;
        out[(size_t)m * kE + e] = s;
      }
}

extern "C" void kernel_launch(void* const* d_in, const int* in_sizes, int n_in,
                              void* d_out, int out_size, void* d_ws,
                              size_t ws_size, hipStream_t stream) {
  const float* x  = (const float*)d_in[0];
  const float* Wq = (const float*)d_in[1];
  const float* bq = (const float*)d_in[2];
  const float* Wk = (const float*)d_in[3];
  const float* bk = (const float*)d_in[4];
  const float* Wv = (const float*)d_in[5];
  const float* bv = (const float*)d_in[6];
  const float* Wo = (const float*)d_in[7];
  const float* bo = (const float*)d_in[8];
  float* out = (float*)d_out;

  char* ws = (char*)d_ws;
  size_t off = 0;
  bf16* inp = (bf16*)(ws + off); off += (size_t)kT * kN * kE * 2;  // 125.8 MB
  bf16* w   = (bf16*)(ws + off); off += (size_t)4 * kE * kE * 2;   //   4.7 MB
  bf16* Qb  = (bf16*)(ws + off); off += (size_t)kN * kE * 2;       //  12.6 MB
  bf16* Kb  = (bf16*)(ws + off); off += (size_t)kN * kE * 2;       //  12.6 MB
  bf16* Vt  = (bf16*)(ws + off); off += (size_t)kN * kE * 2;       //  12.6 MB
  bf16* AO  = (bf16*)(ws + off); off += (size_t)kN * kE * 2;       //  12.6 MB

  k_encode<<<(kN * kE + 255) / 256, 256, 0, stream>>>(x, inp);
  k_cvtw<<<(4 * kE * kE + 255) / 256, 256, 0, stream>>>(Wq, Wk, Wv, Wo, w);
  k_qkv<<<dim3(kN / 128, kE / 64, 3), 256, 0, stream>>>(inp, w, bq, bk, bv,
                                                        Qb, Kb, Vt);
  k_attn<<<dim3(kS / 128, kB * kH), 256, 0, stream>>>(Qb, Kb, Vt, AO);
  k_oproj<<<dim3(kN / 128, kE / 64), 256, 0, stream>>>(
      AO, w + (size_t)3 * kE * kE, bo, out);
}
